// SiSelfAttention_22170621182083
// MI455X (gfx1250) — matrix operand order repaired and checked by reading, not compiled, not run on hardware
//
#include <hip/hip_runtime.h>

#define B_    8
#define T_    2048
#define DM_   1024
#define DH_   128
#define SCALE_ 0.08838834764831845f   // 1/sqrt(128)

typedef __attribute__((ext_vector_type(16))) __bf16 v16bf;
typedef __attribute__((ext_vector_type(8)))  __bf16 v8bf;
typedef __attribute__((ext_vector_type(8)))  float  v8f;

// Build a 16x32 A-fragment's per-lane 16 bf16 values from two contiguous
// 8-element (16B) runs: elements 0..7 = K=kbase..kbase+7, 8..15 = K=kbase+8..+15.
__device__ __forceinline__ v16bf load_ab(const __bf16* p0, const __bf16* p1) {
  v8bf lo = *(const v8bf*)p0;
  v8bf hi = *(const v8bf*)p1;
  v16bf r;
#pragma unroll
  for (int i = 0; i < 8; ++i) { r[i] = lo[i]; r[i + 8] = hi[i]; }
  return r;
}

// ---------------------------------------------------------------------------
// Kernel 1: fused q/k/v projections (fp32 in -> bf16 out), v stored transposed.
// grid = (B*T/16), block = 256 (8 waves; wave w owns h-cols [16w,16w+16) for
// ALL THREE matrices -> each enc A-fragment feeds 3 WMMAs).
// ---------------------------------------------------------------------------
__global__ __launch_bounds__(256) void proj_kernel(
    const float* __restrict__ enc, const float* __restrict__ Wq,
    const float* __restrict__ Wk,  const float* __restrict__ Wv,
    unsigned short* __restrict__ qo, unsigned short* __restrict__ ko,
    unsigned short* __restrict__ vTo)
{
  const int rt   = blockIdx.x;        // 16-row tile over flattened B*T
  const int wv   = threadIdx.x >> 5;
  const int lane = threadIdx.x & 31;
  const int ln15 = lane & 15;
  const int hs   = lane >> 4;
  const int h_base = wv * 16;

  const float* aptr = enc + (size_t)(rt * 16 + ln15) * DM_;   // A: row m = ln15
  const float* bq   = Wq  + (size_t)(h_base + ln15) * DM_;    // B: col n -> W row h
  const float* bk   = Wk  + (size_t)(h_base + ln15) * DM_;
  const float* bv   = Wv  + (size_t)(h_base + ln15) * DM_;

  v8f cq = {}, ck = {}, cv = {};
#pragma unroll 2
  for (int kk = 0; kk < DM_; kk += 32) {
    float af[16];
    const int ka = kk + hs * 16;                // A lane K-base within window
    *(float4*)&af[0]  = *(const float4*)(aptr + ka);
    *(float4*)&af[4]  = *(const float4*)(aptr + ka + 4);
    *(float4*)&af[8]  = *(const float4*)(aptr + ka + 8);
    *(float4*)&af[12] = *(const float4*)(aptr + ka + 12);
    v16bf A;
#pragma unroll
    for (int i = 0; i < 16; ++i) A[i] = (__bf16)af[i];

    const int kb = kk + hs * 16;                // B lane K-base within window
    const float* bps[3] = {bq, bk, bv};
    v8f* accs[3] = {&cq, &ck, &cv};
#pragma unroll
    for (int m = 0; m < 3; ++m) {
      float bf[16];
      *(float4*)&bf[0]  = *(const float4*)(bps[m] + kb);
      *(float4*)&bf[4]  = *(const float4*)(bps[m] + kb + 4);
      *(float4*)&bf[8]  = *(const float4*)(bps[m] + kb + 8);
      *(float4*)&bf[12] = *(const float4*)(bps[m] + kb + 12);
      v16bf Bm;
#pragma unroll
      for (int i = 0; i < 16; ++i) Bm[i] = (__bf16)bf[i];
      *accs[m] = __builtin_amdgcn_wmma_f32_16x16x32_bf16(false, A, false, Bm,
                                                         (short)0, *accs[m],
                                                         false, false);
    }
  }

  const int bt0 = rt * 16;
  // q/k: row-major [B*T][128] bf16. C element r -> (M = hs*8+r, N = ln15).
#pragma unroll
  for (int r = 0; r < 8; ++r) {
    const size_t idx = (size_t)(bt0 + hs * 8 + r) * DH_ + h_base + ln15;
    ((__bf16*)qo)[idx] = (__bf16)cq[r];
    ((__bf16*)ko)[idx] = (__bf16)ck[r];
  }
  // v: transposed [B][128][T] bf16 -> one contiguous 16B store per lane.
  {
    const int b  = bt0 >> 11;            // /T_
    const int t0 = bt0 & (T_ - 1);
    v8bf pk;
#pragma unroll
    for (int r = 0; r < 8; ++r) pk[r] = (__bf16)cv[r];
    *(v8bf*)((__bf16*)vTo + ((size_t)b * DH_ + h_base + ln15) * T_ + t0 + hs * 8) = pk;
  }
}

// ---------------------------------------------------------------------------
// Kernel 2: causal S = qk^T*scale, weights = S^2/||S||, out = weights @ v.
// grid = (T/32, B), block = 256. 32 query rows (two 16-row M-tiles) per block;
// 8 waves split the s-dimension (32-wide chunks). Each k/v B-fragment feeds
// 2 WMMAs (one per M-tile).
// ---------------------------------------------------------------------------
__global__ __launch_bounds__(256) void attn_kernel(
    const unsigned short* __restrict__ qw, const unsigned short* __restrict__ kw,
    const unsigned short* __restrict__ vTw, float* __restrict__ out)
{
  __shared__ __bf16 Pst[8][2][16][32];  // per-wave P=S^2 staging (C -> A layout)
  __shared__ float  accL[32][128];      // cross-wave output accumulator
  __shared__ float  ssL[32][16];        // cross-wave sum-of-squares partials

  const int t_base = blockIdx.x * 32;
  const int b      = blockIdx.y;
  const int wv     = threadIdx.x >> 5;
  const int lane   = threadIdx.x & 31;
  const int ln15   = lane & 15;
  const int hs     = lane >> 4;

  {
    float* p = &accL[0][0];
    for (int i = threadIdx.x; i < 32 * 128; i += 256) p[i] = 0.f;
    float* p2 = &ssL[0][0];
    for (int i = threadIdx.x; i < 32 * 16; i += 256) p2[i] = 0.f;
  }
  __syncthreads();

  const __bf16* qb = (const __bf16*)qw  + ((size_t)b * T_ + t_base) * DH_;
  const __bf16* kb = (const __bf16*)kw  + (size_t)b * T_ * DH_;
  const __bf16* vb = (const __bf16*)vTw + (size_t)b * DH_ * T_;

  // Preload q A-fragments: 2 M-tiles x 4 K=32 steps of the d=128 dot.
  v16bf qa[2][4];
#pragma unroll
  for (int mt = 0; mt < 2; ++mt)
#pragma unroll
    for (int d = 0; d < 4; ++d) {
      const __bf16* p = qb + (size_t)(mt * 16 + ln15) * DH_ + d * 32 + hs * 16;
      qa[mt][d] = load_ab(p, p + 8);
    }

  v8f acc[2][8];
#pragma unroll
  for (int mt = 0; mt < 2; ++mt)
#pragma unroll
    for (int i = 0; i < 8; ++i) acc[mt][i] = (v8f){};
  float ss[2][8];
#pragma unroll
  for (int mt = 0; mt < 2; ++mt)
#pragma unroll
    for (int r = 0; r < 8; ++r) ss[mt][r] = 0.f;

  const int t_max = t_base + 31;
  const int ns2   = (t_max >> 5) + 1;          // # of 32-wide s chunks (causal)

  for (int s2 = wv; s2 < ns2; s2 += 8) {
    const int s2b = s2 * 32;
#pragma unroll
    for (int hlf = 0; hlf < 2; ++hlf) {
      const int sb = s2b + hlf * 16;
      v8f sres[2] = {{}, {}};
#pragma unroll
      for (int d = 0; d < 4; ++d) {   // S tiles: q(2x16x128) @ k^T(128x16)
        const v16bf bfrag =
            *(const v16bf*)(kb + (size_t)(sb + ln15) * DH_ + d * 32 + hs * 16);
        sres[0] = __builtin_amdgcn_wmma_f32_16x16x32_bf16(false, qa[0][d], false,
                      bfrag, (short)0, sres[0], false, false);
        sres[1] = __builtin_amdgcn_wmma_f32_16x16x32_bf16(false, qa[1][d], false,
                      bfrag, (short)0, sres[1], false, false);
      }
      // mask (multiplicative causal) -> scale -> square; track row sum squares
      const int scol = sb + ln15;
#pragma unroll
      for (int mt = 0; mt < 2; ++mt) {
        const int trow = t_base + mt * 16 + hs * 8;
#pragma unroll
        for (int r = 0; r < 8; ++r) {
          float sv = sres[mt][r] * SCALE_;
          sv = (scol <= trow + r) ? sv : 0.f;
          float pw = sv * sv;
          ss[mt][r] += pw;
          Pst[wv][mt][hs * 8 + r][hlf * 16 + ln15] = (__bf16)pw;
        }
      }
    }
    // Re-read P in A-fragment layout (same wave; LDS is in-order).
    v16bf pa[2];
#pragma unroll
    for (int mt = 0; mt < 2; ++mt)
      pa[mt] = load_ab(&Pst[wv][mt][ln15][hs * 16], &Pst[wv][mt][ln15][hs * 16 + 8]);
#pragma unroll
    for (int ht = 0; ht < 8; ++ht) {  // P(16x32) @ v(32x16) for all 128 cols
      const v16bf vfrag =
          *(const v16bf*)(vb + (size_t)(ht * 16 + ln15) * T_ + s2b + hs * 16);
      acc[0][ht] = __builtin_amdgcn_wmma_f32_16x16x32_bf16(false, pa[0], false,
                       vfrag, (short)0, acc[0][ht], false, false);
      acc[1][ht] = __builtin_amdgcn_wmma_f32_16x16x32_bf16(false, pa[1], false,
                       vfrag, (short)0, acc[1][ht], false, false);
    }
  }

  // Cross-wave reduction via LDS float atomics (ds_add_f32).
#pragma unroll
  for (int mt = 0; mt < 2; ++mt) {
#pragma unroll
    for (int ht = 0; ht < 8; ++ht)
#pragma unroll
      for (int r = 0; r < 8; ++r)
        atomicAdd(&accL[mt * 16 + hs * 8 + r][ht * 16 + ln15], acc[mt][ht][r]);
#pragma unroll
    for (int r = 0; r < 8; ++r)
      atomicAdd(&ssL[mt * 16 + hs * 8 + r][ln15], ss[mt][r]);
  }
  __syncthreads();

  // Finalize: out = acc * rsqrt(sum_s S^2). 256 threads x 16 elements.
  const int M  = threadIdx.x >> 3;          // 0..31
  const int hg = (threadIdx.x & 7) * 16;    // 0,16,...,112
  float ssum = 0.f;
#pragma unroll
  for (int n = 0; n < 16; ++n) ssum += ssL[M][n];
  const float rn = (ssum > 0.f) ? rsqrtf(ssum) : 0.f;
  float* op = out + ((size_t)b * T_ + t_base + M) * DH_ + hg;
#pragma unroll
  for (int j = 0; j < 16; ++j) op[j] = accL[M][hg + j] * rn;
}

// ---------------------------------------------------------------------------
extern "C" void kernel_launch(void* const* d_in, const int* in_sizes, int n_in,
                              void* d_out, int out_size, void* d_ws, size_t ws_size,
                              hipStream_t stream) {
  (void)in_sizes; (void)n_in; (void)out_size; (void)ws_size;
  const float* enc = (const float*)d_in[0];
  const float* Wq  = (const float*)d_in[1];
  const float* Wk  = (const float*)d_in[2];
  const float* Wv  = (const float*)d_in[3];
  float* out = (float*)d_out;

  // Workspace: q[B*T*128] bf16, k[B*T*128] bf16, vT[B*128*T] bf16 = 12 MB total.
  unsigned short* ws = (unsigned short*)d_ws;
  unsigned short* qws = ws;
  unsigned short* kws = qws + (size_t)B_ * T_ * DH_;
  unsigned short* vTs = kws + (size_t)B_ * T_ * DH_;

  proj_kernel<<<dim3((B_ * T_) / 16), 256, 0, stream>>>(
      enc, Wq, Wk, Wv, qws, kws, vTs);
  attn_kernel<<<dim3(T_ / 32, B_), 256, 0, stream>>>(qws, kws, vTs, out);
}
